// OptimizedFireTVNeuralCDE_850403525425
// MI455X (gfx1250) — matrix operand order repaired and checked by reading, not compiled, not run on hardware
//
#include <hip/hip_runtime.h>
#include <hip/hip_bf16.h>

// ---------------------------------------------------------------------------
// FireTV NeuralCDE on MI455X (gfx1250, wave32, WMMA bf16)
// B=256, S=33, D=42, H=128, HFF=256, T=20
// All activations flow as bf16 row-major (K-contiguous == WMMA pair format).
// ---------------------------------------------------------------------------

#define BB    256
#define SS    33
#define DD    42
#define HH    128
#define HFF_  256
#define TT    20
#define NSTEP 32            // S-1
#define EHN   (HFF_ * HH)   // 32768 (Weff inner size, n = h*HFF + e)

typedef __bf16 bf16_t;
typedef __attribute__((ext_vector_type(16))) __bf16 v16bf;
typedef __attribute__((ext_vector_type(8)))  float  v8f;

static __device__ __forceinline__ unsigned pack2bf(float a, float b) {
    bf16_t ha = (bf16_t)a, hb = (bf16_t)b;               // RNE converts
    unsigned short ua = __builtin_bit_cast(unsigned short, ha);
    unsigned short ub = __builtin_bit_cast(unsigned short, hb);
    return (unsigned)ua | ((unsigned)ub << 16);
}
static __device__ __forceinline__ float bflo(unsigned u) {
    return __builtin_bit_cast(float, u << 16);
}
static __device__ __forceinline__ float bfhi(unsigned u) {
    return __builtin_bit_cast(float, u & 0xffff0000u);
}

// ---------------------------------------------------------------------------
// WMMA GEMM:  C[bz] = act( (A[bz] @ Wp[bz]) * scale[n] + shift[bz][n] )
//   A  : bf16 row-major, M=256 rows, lda elems/row, K padded to mult of 32
//        (tail rows of Wp are zero so garbage A in [K,Kpad) is harmless).
//   Wp : bf16 pair-interleaved uints, (Kpad/2) x Npad:  Wp[p][n]={W[2p][n],W[2p+1][n]}
//   Cf : optional fp32 out; Cb: optional bf16 out (same ldc/cbatch).
//   Block = 128 thr = 4 waves; tile 16(M) x 64(N); double-buffered LDS.
//   act: 0=none 1=gelu(exact).  blockIdx.z batches (strides in elems).
// ---------------------------------------------------------------------------
__launch_bounds__(128)
__global__ void wmma_gemm_kernel(const bf16_t* __restrict__ A, int Kpad, int lda, long abatch,
                                 const unsigned* __restrict__ Wp, int Npad, int Nvalid, long wbatch,
                                 const float* __restrict__ scale,
                                 const float* __restrict__ shift, int shbatch,
                                 int act,
                                 float* __restrict__ Cf, bf16_t* __restrict__ Cb,
                                 int ldc, long cbatch)
{
    __shared__ unsigned As2[2][16][16];   // packed bf16 pairs along K
    __shared__ unsigned Bs2[2][16][64];   // packed bf16 pairs along K

    const int tid  = threadIdx.x;
    const int lane = tid & 31;
    const int wave = tid >> 5;
    const int m0   = blockIdx.y * 16;
    const int nb0  = blockIdx.x * 64;
    const int bz   = blockIdx.z;

    const bf16_t*   Ab = A  + (size_t)bz * abatch;
    const unsigned* Wb = Wp + (size_t)bz * wbatch;

    const int r16 = tid >> 3;       // 0..15
    const int c8  = tid & 7;        // 0..7

    auto stageA = [&](int kc, int buf) {
        // 16 rows x 32 bf16: one uint2 (4 bf16 = 2 K-pairs) per thread, no cvt
        const uint2 u = *(const uint2*)(Ab + (size_t)(m0 + r16) * lda + kc + c8 * 4);
        As2[buf][r16][c8 * 2]     = u.x;
        As2[buf][r16][c8 * 2 + 1] = u.y;
    };
    auto stageB = [&](int kc, int buf) {
        // 16 pair-rows x 64 uints: two uint4 per thread, fully coalesced
        const unsigned* src = Wb + (size_t)((kc >> 1) + r16) * Npad + nb0 + c8 * 8;
        uint4 u0 = *(const uint4*)src;
        uint4 u1 = *(const uint4*)(src + 4);
        *(uint4*)&Bs2[buf][r16][c8 * 8]     = u0;
        *(uint4*)&Bs2[buf][r16][c8 * 8 + 4] = u1;
    };

    v8f acc = {};
    const int mrow  = lane & 15;
    const int khalf = lane >> 4;
    const int nl    = wave * 16 + (lane & 15);

    int cur = 0;
    stageA(0, 0);
    stageB(0, 0);
    __syncthreads();

    for (int kc = 0; kc < Kpad; kc += 32) {
        const int nxt = cur ^ 1;
        if (kc + 32 < Kpad) {            // uniform branch
            stageA(kc + 32, nxt);
            stageB(kc + 32, nxt);
            if (kc + 64 < Kpad)          // speculative prefetch of chunk after next
                __builtin_prefetch(Wb + (size_t)((kc >> 1) + 32) * Npad + nb0, 0, 0);
        }
        // fragment build straight into WMMA register images (ISA 7.12.2 layouts)
        union { v16bf v; unsigned u[8]; } fa, fb;
#pragma unroll
        for (int q = 0; q < 8; ++q) {
            const int pA = ((q >> 2) << 3) + (khalf << 2) + (q & 3);  // A 16x32 pairs
            fa.u[q] = As2[cur][mrow][pA];
            fb.u[q] = Bs2[cur][pA][nl];                 // B 32x16 pairs
        }
        // EXEC full here (uniform control flow) as WMMA requires.
        acc = __builtin_amdgcn_wmma_f32_16x16x32_bf16(
                  false, fa.v, false, fb.v, (short)0, acc, false, false);
        __syncthreads();
        cur = nxt;
    }

    // ---- epilogue: fused affine (BN+bias folded) + activation ----
    const int ncol = nb0 + wave * 16 + (lane & 15);
    if (ncol < Nvalid) {
        const float sc = scale ? scale[ncol] : 1.f;
        const float sh = shift ? shift[(size_t)bz * shbatch + ncol] : 0.f;
#pragma unroll
        for (int r = 0; r < 8; ++r) {
            const int mr = m0 + r + (khalf << 3);
            float v = acc[r] * sc + sh;
            if (act == 1) v = 0.5f * v * (1.f + erff(v * 0.70710678118654752f));
            if (Cf) (Cf + (size_t)bz * cbatch)[(size_t)mr * ldc + ncol] = v;
            if (Cb) (Cb + (size_t)bz * cbatch)[(size_t)mr * ldc + ncol] = (bf16_t)v;
        }
    }
}

// ---------------------------------------------------------------------------
// Packing / prep kernels (one-time)
// ---------------------------------------------------------------------------
// Generic pair-interleave pack: dst[p*Npad + n] = pack(src[2p][n], src[2p+1][n])
// Zero-fill for k>=K or n>=N. Batched over blockIdx.y.
__global__ void pack_pairs_kernel(const float* __restrict__ src, int K, int Kpad, int N, int Npad,
                                  unsigned* __restrict__ dst, long sstride, long dstride) {
    const int total = (Kpad >> 1) * Npad;
    int i = blockIdx.x * blockDim.x + threadIdx.x;
    if (i >= total) return;
    const float* s = src + (size_t)blockIdx.y * sstride;
    unsigned* d    = dst + (size_t)blockIdx.y * dstride;
    int p = i / Npad, n = i % Npad;
    int k0 = 2 * p, k1 = 2 * p + 1;
    float v0 = (k0 < K && n < N) ? s[(size_t)k0 * N + n] : 0.f;
    float v1 = (k1 < K && n < N) ? s[(size_t)k1 * N + n] : 0.f;
    d[i] = pack2bf(v0, v1);
}

// W3 pack with transpose + n-remap: n = h*HFF + e ; K dim = d (42 -> pad 64)
// dst[p*EHN + h*HFF + e] = pack(w3[e][h*D+2p], w3[e][h*D+2p+1])
__global__ void pack_w3t_kernel(const float* __restrict__ w3, unsigned* __restrict__ dst) {
    const int total = 32 * EHN;     // Kpad/2 = 32
    int i = blockIdx.x * blockDim.x + threadIdx.x;
    if (i >= total) return;
    int p = i / EHN, n = i % EHN;
    int h = n / HFF_, e = n % HFF_;
    int d0 = 2 * p, d1 = 2 * p + 1;
    float v0 = (d0 < DD) ? w3[(size_t)e * (HH * DD) + h * DD + d0] : 0.f;
    float v1 = (d1 < DD) ? w3[(size_t)e * (HH * DD) + h * DD + d1] : 0.f;
    dst[i] = pack2bf(v0, v1);
}

// b3r pack: dst[p*HH + h] = pack(b3[h*D+2p], b3[h*D+2p+1])
__global__ void pack_b3r_kernel(const float* __restrict__ b3, unsigned* __restrict__ dst) {
    const int total = 32 * HH;
    int i = blockIdx.x * blockDim.x + threadIdx.x;
    if (i >= total) return;
    int p = i / HH, h = i % HH;
    int d0 = 2 * p, d1 = 2 * p + 1;
    float v0 = (d0 < DD) ? b3[h * DD + d0] : 0.f;
    float v1 = (d1 < DD) ? b3[h * DD + d1] : 0.f;
    dst[i] = pack2bf(v0, v1);
}

// Fold BN into affine epilogue: sc = g/sqrt(1+eps), sh = b*sc + be
__global__ void eff_prep_kernel(const float* __restrict__ g, const float* __restrict__ b,
                                const float* __restrict__ be,
                                float* __restrict__ sc, float* __restrict__ sh, int n) {
    int i = blockIdx.x * blockDim.x + threadIdx.x;
    if (i >= n) return;
    float s = g[i] * rsqrtf(1.f + 1e-5f);
    sc[i] = s;
    sh[i] = b[i] * s + be[i];
}

// x0b[b][c] = bf16( c<D ? path[b][0][c] : 0 )   (B x 64)
__global__ void x0_prep_kernel(const float* __restrict__ path, bf16_t* __restrict__ x0b) {
    int i = blockIdx.x * blockDim.x + threadIdx.x;
    if (i >= BB * 64) return;
    int b = i >> 6, c = i & 63;
    x0b[i] = (bf16_t)((c < DD) ? path[(size_t)b * SS * DD + c] : 0.f);
}

// dt[s] = t[s+1]-t[s];  dXdtb[s][b][c] = bf16( c<D ? (X[b][s+1][c]-X[b][s][c])/dt[s] : 0 )
__global__ void dxdt_prep_kernel(const float* __restrict__ path, const float* __restrict__ ts,
                                 bf16_t* __restrict__ dXdtb, float* __restrict__ dt) {
    int i = blockIdx.x * blockDim.x + threadIdx.x;
    if (i < NSTEP) dt[i] = ts[i + 1] - ts[i];
    if (i >= NSTEP * BB * 64) return;
    int s = i / (BB * 64);
    int r = i % (BB * 64);
    int b = r >> 6, c = r & 63;
    float v = 0.f;
    if (c < DD) {
        float dts = ts[s + 1] - ts[s];
        v = (path[(size_t)b * SS * DD + (s + 1) * DD + c] -
             path[(size_t)b * SS * DD + s * DD + c]) / dts;
    }
    dXdtb[i] = (bf16_t)v;
}

// ---------------------------------------------------------------------------
// g-dot + fused RK4 axpy:
//   k[b,h]  = sum_e y[b,e]*Weff[b,h,e] + bterm[b,h]     (y, Weff bf16)
//   ztb[b,h] = bf16( zz[b,h] + c*dt[s]*k[b,h] )         (next GEMM A operand)
// ---------------------------------------------------------------------------
__launch_bounds__(128)
__global__ void gdot_axpy_kernel(const bf16_t* __restrict__ y, const bf16_t* __restrict__ Weff,
                                 const float* __restrict__ bterm, const float* __restrict__ zz,
                                 const float* __restrict__ dt, int s, float c,
                                 float* __restrict__ kout, bf16_t* __restrict__ ztb) {
    const int b = blockIdx.x;
    const int h = threadIdx.x;       // 128 threads
    __shared__ float ys[HFF_];
    {
        unsigned u = ((const unsigned*)(y + (size_t)b * HFF_))[h];  // 2 bf16 per thread
        ys[2 * h]     = bflo(u);
        ys[2 * h + 1] = bfhi(u);
    }
    __syncthreads();
    const uint4* wb = (const uint4*)(Weff + (size_t)b * EHN + (size_t)h * HFF_);
    float acc = 0.f;
#pragma unroll
    for (int i = 0; i < 32; ++i) {   // 32 x uint4 = 256 bf16
        uint4 u = wb[i];
        int e = i * 8;
        acc = fmaf(ys[e + 0], bflo(u.x), acc);
        acc = fmaf(ys[e + 1], bfhi(u.x), acc);
        acc = fmaf(ys[e + 2], bflo(u.y), acc);
        acc = fmaf(ys[e + 3], bfhi(u.y), acc);
        acc = fmaf(ys[e + 4], bflo(u.z), acc);
        acc = fmaf(ys[e + 5], bfhi(u.z), acc);
        acc = fmaf(ys[e + 6], bflo(u.w), acc);
        acc = fmaf(ys[e + 7], bfhi(u.w), acc);
    }
    const int idx = b * HH + h;
    const float k = acc + bterm[idx];
    kout[idx] = k;
    ztb[idx] = (bf16_t)(zz[idx] + c * dt[s] * k);
}

// z += dt[s]/6 * (k1 + 2k2 + 2k3 + k4) ; zb = bf16(z)
__global__ void rk4_update_kernel(float* __restrict__ z, bf16_t* __restrict__ zb,
                                  const float* __restrict__ k1, const float* __restrict__ k2,
                                  const float* __restrict__ k3, const float* __restrict__ k4,
                                  const float* __restrict__ dt, int s, int n) {
    int i = blockIdx.x * blockDim.x + threadIdx.x;
    if (i >= n) return;
    float v = z[i] + (dt[s] * (1.f / 6.f)) * (k1[i] + 2.f * k2[i] + 2.f * k3[i] + k4[i]);
    z[i] = v;
    zb[i] = (bf16_t)v;
}

// out[b*T+t] = sigmoid( h2[t][b][:] . dw3[t] + db3[t] )
__global__ void dec3_kernel(const float* __restrict__ h2, const float* __restrict__ dw3,
                            const float* __restrict__ db3, float* __restrict__ out) {
    int i = blockIdx.x * blockDim.x + threadIdx.x;
    if (i >= BB * TT) return;
    int b = i / TT, t = i % TT;
    const float4* hp = (const float4*)(h2 + ((size_t)t * BB + b) * 32);
    const float4* wp = (const float4*)(dw3 + t * 32);
    float acc = db3[t];
#pragma unroll
    for (int q = 0; q < 8; ++q) {
        float4 hv = hp[q], wv = wp[q];
        acc = fmaf(hv.x, wv.x, fmaf(hv.y, wv.y, fmaf(hv.z, wv.z, fmaf(hv.w, wv.w, acc))));
    }
    out[i] = 1.f / (1.f + __expf(-acc));
}

// ---------------------------------------------------------------------------
// Host launcher
// ---------------------------------------------------------------------------
extern "C" void kernel_launch(void* const* d_in, const int* in_sizes, int n_in,
                              void* d_out, int out_size, void* d_ws, size_t ws_size,
                              hipStream_t stream) {
    (void)in_sizes; (void)n_in; (void)out_size; (void)ws_size;
    const float* path    = (const float*)d_in[0];
    const float* ts      = (const float*)d_in[1];
    const float* enc_w1  = (const float*)d_in[2];
    const float* enc_b1  = (const float*)d_in[3];
    const float* enc_g1  = (const float*)d_in[4];
    const float* enc_be1 = (const float*)d_in[5];
    const float* enc_w2  = (const float*)d_in[6];
    const float* enc_b2  = (const float*)d_in[7];
    const float* enc_g2  = (const float*)d_in[8];
    const float* enc_be2 = (const float*)d_in[9];
    const float* vf_w0   = (const float*)d_in[10];
    const float* vf_b0   = (const float*)d_in[11];
    const float* vf_g0   = (const float*)d_in[12];
    const float* vf_be0  = (const float*)d_in[13];
    const float* vf_w1   = (const float*)d_in[14];
    const float* vf_b1   = (const float*)d_in[15];
    const float* vf_g1   = (const float*)d_in[16];
    const float* vf_be1  = (const float*)d_in[17];
    const float* vf_w2   = (const float*)d_in[18];
    const float* vf_b2   = (const float*)d_in[19];
    const float* vf_g2   = (const float*)d_in[20];
    const float* vf_be2  = (const float*)d_in[21];
    const float* vf_w3   = (const float*)d_in[22];
    const float* vf_b3   = (const float*)d_in[23];
    const float* wv      = (const float*)d_in[24];
    const float* bv      = (const float*)d_in[25];
    const float* wo      = (const float*)d_in[26];
    const float* bo      = (const float*)d_in[27];
    const float* dw1     = (const float*)d_in[28];
    const float* db1     = (const float*)d_in[29];
    const float* dw2     = (const float*)d_in[30];
    const float* db2     = (const float*)d_in[31];
    const float* dw3     = (const float*)d_in[32];
    const float* db3     = (const float*)d_in[33];
    float* out = (float*)d_out;

    // --- workspace bump allocator (256B aligned) ---
    char* wsb = (char*)d_ws;
    size_t off = 0;
    auto alloc = [&](size_t bytes) -> void* {
        off = (off + 255) & ~(size_t)255;
        void* p = wsb + off;
        off += bytes;
        return p;
    };
    auto fbuf = [&](size_t n) { return (float*)alloc(n * sizeof(float)); };
    auto ubuf = [&](size_t n) { return (unsigned*)alloc(n * sizeof(unsigned)); };
    auto bbuf = [&](size_t n) { return (bf16_t*)alloc(n * sizeof(bf16_t)); };

    // packed weights (uint = 2 bf16 along K)
    unsigned* enc_w1p = ubuf(32 * 128);            // Kpad 64
    unsigned* enc_w2p = ubuf(64 * 128);
    unsigned* w0p     = ubuf(64 * 256);
    unsigned* w1p     = ubuf(128 * 256);
    unsigned* w2p     = ubuf(128 * 256);
    unsigned* w3p     = ubuf((size_t)32 * EHN);    // Kpad 64, n = h*HFF+e
    unsigned* b3rp    = ubuf(32 * 128);
    unsigned* wvp     = ubuf(64 * 128);
    unsigned* wop     = ubuf(64 * 128);
    unsigned* dw1p    = ubuf((size_t)TT * 64 * 64);
    unsigned* dw2p    = ubuf((size_t)TT * 32 * 64);    // Npad 64
    bf16_t*   weffb   = bbuf((size_t)BB * EHN);        // 16 MB, L2-resident

    // bf16 activations (GEMM A operands)
    bf16_t* x0b   = bbuf(BB * 64);
    bf16_t* dXdtb = bbuf((size_t)NSTEP * BB * 64);
    bf16_t* zzb   = bbuf(BB * HH);
    bf16_t* ztb   = bbuf(BB * HH);
    bf16_t* y0b   = bbuf(BB * HFF_);
    bf16_t* y1b   = bbuf(BB * HFF_);
    bf16_t* y2b   = bbuf(BB * HFF_);
    bf16_t* attnb = bbuf(BB * HH);
    bf16_t* attdb = bbuf(BB * HH);
    bf16_t* h1b   = bbuf((size_t)TT * BB * 64);

    // fp32 scratch
    float* sc1 = fbuf(HH);   float* sh1 = fbuf(HH);
    float* sc2 = fbuf(HH);   float* sh2 = fbuf(HH);
    float* sl0 = fbuf(HFF_); float* hl0 = fbuf(HFF_);
    float* sl1 = fbuf(HFF_); float* hl1 = fbuf(HFF_);
    float* sl2 = fbuf(HFF_); float* hl2 = fbuf(HFF_);
    float* dt    = fbuf(NSTEP);
    float* zz    = fbuf(BB * HH);
    float* k1    = fbuf(BB * HH);
    float* k2    = fbuf(BB * HH);
    float* k3    = fbuf(BB * HH);
    float* k4    = fbuf(BB * HH);
    float* bterm = fbuf(BB * HH);
    float* h2    = fbuf((size_t)TT * BB * 32);

    // --- one-time packing / folding ---
    auto pack = [&](const float* s, int K, int Kpad, int N, int Npad, unsigned* d,
                    int batch = 1, long sstride = 0, long dstride = 0) {
        int total = (Kpad >> 1) * Npad;
        pack_pairs_kernel<<<dim3((total + 255) / 256, batch), 256, 0, stream>>>(
            s, K, Kpad, N, Npad, d, sstride, dstride);
    };
    pack(enc_w1, DD, 64, HH, HH, enc_w1p);
    pack(enc_w2, HH, HH, HH, HH, enc_w2p);
    pack(vf_w0, HH, HH, HFF_, HFF_, w0p);
    pack(vf_w1, HFF_, HFF_, HFF_, HFF_, w1p);
    pack(vf_w2, HFF_, HFF_, HFF_, HFF_, w2p);
    pack(wv, HH, HH, HH, HH, wvp);
    pack(wo, HH, HH, HH, HH, wop);
    pack(dw1, HH, HH, 64, 64, dw1p, TT, HH * 64, 64 * 64);
    pack(dw2, 64, 64, 32, 64, dw2p, TT, 64 * 32, 32 * 64);
    pack_w3t_kernel<<<(32 * EHN + 255) / 256, 256, 0, stream>>>(vf_w3, w3p);
    pack_b3r_kernel<<<(32 * HH + 255) / 256, 256, 0, stream>>>(vf_b3, b3rp);
    eff_prep_kernel<<<1, HH,   0, stream>>>(enc_g1, enc_b1, enc_be1, sc1, sh1, HH);
    eff_prep_kernel<<<1, HH,   0, stream>>>(enc_g2, enc_b2, enc_be2, sc2, sh2, HH);
    eff_prep_kernel<<<1, HFF_, 0, stream>>>(vf_g0, vf_b0, vf_be0, sl0, hl0, HFF_);
    eff_prep_kernel<<<1, HFF_, 0, stream>>>(vf_g1, vf_b1, vf_be1, sl1, hl1, HFF_);
    eff_prep_kernel<<<1, HFF_, 0, stream>>>(vf_g2, vf_b2, vf_be2, sl2, hl2, HFF_);
    x0_prep_kernel<<<(BB * 64 + 255) / 256, 256, 0, stream>>>(path, x0b);
    dxdt_prep_kernel<<<(NSTEP * BB * 64 + 255) / 256, 256, 0, stream>>>(path, ts, dXdtb, dt);

    // GEMM launch helper (M fixed = 256)
    auto gemm = [&](const bf16_t* A, int Kpad, int lda, long abatch,
                    const unsigned* W, int Npad, int Nvalid, long wbatch,
                    const float* scale, const float* shift, int shbatch,
                    int act, float* Cf, bf16_t* Cb, int ldc, long cbatch, int batch = 1) {
        wmma_gemm_kernel<<<dim3(Npad / 64, BB / 16, batch), 128, 0, stream>>>(
            A, Kpad, lda, abatch, W, Npad, Nvalid, wbatch,
            scale, shift, shbatch, act, Cf, Cb, ldc, cbatch);
    };

    // --- encoder: z0 fp32 master + bf16 mirror ---
    gemm(x0b, 64, 64, 0, enc_w1p, HH, HH, 0, sc1, sh1, 0, 1, nullptr, y0b, HH, 0);
    gemm(y0b, HH, HH, 0, enc_w2p, HH, HH, 0, sc2, sh2, 0, 0, zz, zzb, HH, 0);

    const int nz = BB * HH;
    auto eval_g = [&](const bf16_t* zin, float* kout, float c, int s) {
        gemm(zin, HH, HH, 0, w0p, HFF_, HFF_, 0, sl0, hl0, 0, 1, nullptr, y0b, HFF_, 0);
        gemm(y0b, HFF_, HFF_, 0, w1p, HFF_, HFF_, 0, sl1, hl1, 0, 1, nullptr, y1b, HFF_, 0);
        gemm(y1b, HFF_, HFF_, 0, w2p, HFF_, HFF_, 0, sl2, hl2, 0, 1, nullptr, y2b, HFF_, 0);
        gdot_axpy_kernel<<<BB, 128, 0, stream>>>(y2b, weffb, bterm, zz, dt, s, c, kout, ztb);
    };

    // --- 32 RK4 steps; Weff + bterm hoisted (dxdt shared by all 4 stages) ---
    for (int s = 0; s < NSTEP; ++s) {
        const bf16_t* dxs = dXdtb + (size_t)s * BB * 64;   // [b][64] padded
        gemm(dxs, 64, 64, 0, w3p, EHN, EHN, 0, nullptr, nullptr, 0, 0, nullptr, weffb, EHN, 0);
        gemm(dxs, 64, 64, 0, b3rp, HH, HH, 0, nullptr, nullptr, 0, 0, bterm, nullptr, HH, 0);

        eval_g(zzb, k1, 0.5f, s);
        eval_g(ztb, k2, 0.5f, s);
        eval_g(ztb, k3, 1.0f, s);
        eval_g(ztb, k4, 0.0f, s);   // ztb write harmless
        rk4_update_kernel<<<(nz + 255) / 256, 256, 0, stream>>>(zz, zzb, k1, k2, k3, k4, dt, s, nz);
    }

    // --- attention head: attended = (zT@wv + bv)@wo + bo ---
    gemm(zzb,   HH, HH, 0, wvp, HH, HH, 0, nullptr, bv, 0, 0, nullptr, attnb, HH, 0);
    gemm(attnb, HH, HH, 0, wop, HH, HH, 0, nullptr, bo, 0, 0, nullptr, attdb, HH, 0);

    // --- per-task decoders, batched over blockIdx.z = t ---
    gemm(attdb, HH, HH, 0, dw1p, 64, 64, 64 * 64, nullptr, db1, 64, 1,
         nullptr, h1b, 64, (long)BB * 64, TT);
    gemm(h1b, 64, 64, (long)BB * 64, dw2p, 64, 32, 32 * 64, nullptr, db2, 32, 1,
         h2, nullptr, 32, (long)BB * 32, TT);
    dec3_kernel<<<(BB * TT + 255) / 256, 256, 0, stream>>>(h2, dw3, db3, out);
}
